// ContentAttention_56118042689955
// MI455X (gfx1250) — matrix operand order repaired and checked by reading, not compiled, not run on hardware
//
#include <hip/hip_runtime.h>
#include <hip/hip_bf16.h>

// CDNA5 / gfx1250, wave32. bf16 WMMA pipeline for the fused attention block.
// Round 4: keep the cur/nxt double-buffered B pipeline, but split 8-tile
// stages into two 4-tile passes (acc32+cur32+nxt32 VGPRs) and relax the
// occupancy target so nothing spills to scratch.

typedef __attribute__((ext_vector_type(16))) __bf16 bf16x16;
typedef __attribute__((ext_vector_type(8)))  __bf16 bf16x8;
typedef __attribute__((ext_vector_type(8)))  float  f32x8;

#define DEV __device__ __forceinline__

static constexpr int Dm  = 512;   // model dim
static constexpr int Tk  = 256;   // keys per batch
// padded LDS pitches (avoid 64-bank conflicts on A-fragment ds_load_b128)
static constexpr int LQ = 520;    // bf16 pitch for 512-wide tiles
static constexpr int LS = 260;    // f32 pitch for 256-wide scores
static constexpr int LP = 264;    // bf16 pitch for 256-wide probabilities

DEV f32x8 fzero() { f32x8 z = {0.f,0.f,0.f,0.f,0.f,0.f,0.f,0.f}; return z; }

DEV f32x8 wmma_bf16(bf16x16 a, bf16x16 b, f32x8 c) {
  // D = A(16x32) x B(32x16) + C, f32 accumulate
  return __builtin_amdgcn_wmma_f32_16x16x32_bf16(false, a, false, b, (short)0, c, false, false);
}

// A fragment 16x32 (MxK): lane holds row M=lane&15; K-halves split by lane>=16.
// Two contiguous 8-element (16B) chunks per lane: [kb, kb+8) and [kb+8, kb+16).
DEV bf16x16 load_a(const __bf16* base, int ldk, int m0, int k0, int lane) {
  int row = m0 + (lane & 15);
  int kb  = k0 + ((lane & 16) ? 16 : 0);
  const __bf16* p = base + row * ldk + kb;
  union { bf16x16 v; bf16x8 h[2]; } u;
  u.h[0] = *(const bf16x8*)(p);
  u.h[1] = *(const bf16x8*)(p + 8);
  return u.v;
}

// B fragment 32x16 (KxN) where element B[k][n] lives at base[n*ld + k]
// (column of B contiguous in k). One 32B contiguous read per lane.
DEV bf16x16 load_b(const __bf16* base, int ld, int n0, int k0, int lane) {
  int n  = n0 + (lane & 15);
  int kb = k0 + ((lane & 16) ? 16 : 0);
  return *(const bf16x16*)(base + (size_t)n * ld + kb);
}

// Software-pipelined 4-tile GEMM pass: acc[4] += A_lds(16 x KTOT) x B_glb tiles.
// B fragments double-buffered across k-steps; loads for step k+1 issued before
// the WMMA chain of step k. 4 tiles keeps the live set ~130 VGPRs (no spills).
template <int KTOT>
DEV void gemm_pass4(const __bf16* lds_a, int lda, int mi,
                    const __bf16* gB, int ldb, int n_base,
                    int lane, f32x8* acc) {
  bf16x16 cur[4], nxt[4];
#pragma unroll
  for (int t = 0; t < 4; ++t) cur[t] = load_b(gB, ldb, n_base + t * 16, 0, lane);
#pragma unroll 1
  for (int k = 0; k < KTOT; k += 64) {
#pragma unroll
    for (int t = 0; t < 4; ++t) nxt[t] = load_b(gB, ldb, n_base + t * 16, k + 32, lane);
    bf16x16 a0 = load_a(lds_a, lda, mi, k, lane);
#pragma unroll
    for (int t = 0; t < 4; ++t) acc[t] = wmma_bf16(a0, cur[t], acc[t]);
    if (k + 64 < KTOT) {
#pragma unroll
      for (int t = 0; t < 4; ++t) cur[t] = load_b(gB, ldb, n_base + t * 16, k + 64, lane);
    }
    bf16x16 a1 = load_a(lds_a, lda, mi, k + 32, lane);
#pragma unroll
    for (int t = 0; t < 4; ++t) acc[t] = wmma_bf16(a1, nxt[t], acc[t]);
  }
}

// ---------------- prep kernels ----------------

__global__ void cvt_w_kernel(const float* __restrict__ Wq, const float* __restrict__ Wk,
                             __bf16* __restrict__ Wqb, __bf16* __restrict__ Wkb) {
  int i = blockIdx.x * blockDim.x + threadIdx.x;
  if (i < Dm * Dm) { Wqb[i] = (__bf16)Wq[i]; Wkb[i] = (__bf16)Wk[i]; }
}

__global__ void transpose_v_kernel(const float* __restrict__ V, __bf16* __restrict__ vT) {
  // V: [8, 256, 512] f32 -> vT: [8, 512, 256] bf16
  int i = blockIdx.x * blockDim.x + threadIdx.x;
  if (i >= 8 * Tk * Dm) return;
  int d = i & (Dm - 1);
  int t = (i >> 9) & (Tk - 1);
  int b = i >> 17;
  vT[((size_t)b * Dm + d) * Tk + t] = (__bf16)V[i];
}

// ---------------- K projection: kproj = key @ Wk^T + bk (bf16 out) ----------------
// 32 rows per WG, 8 wave32 per WG. wave w: M-half = (w>>2)*16, 8 N-tiles of 16.
__global__ void __launch_bounds__(256, 1)
proj_kernel(const float* __restrict__ X,     // [M, 512] f32 rows
            const __bf16* __restrict__ Wb,   // [512,512] bf16, row-major [n][k]
            const float* __restrict__ bias,  // [512]
            __bf16* __restrict__ Y) {        // [M, 512] bf16
  extern __shared__ char smem[];
  __bf16* xs = (__bf16*)smem;                 // [32][LQ]
  const int tid = threadIdx.x, lane = tid & 31, wave = tid >> 5;
  const size_t m_base = (size_t)blockIdx.x * 32;

  for (int i = tid * 4; i < 32 * Dm; i += 256 * 4) {
    float4 v = *(const float4*)(X + m_base * Dm + i);
    int r = i >> 9, c = i & (Dm - 1);
    __bf16* q = xs + r * LQ + c;
    q[0] = (__bf16)v.x; q[1] = (__bf16)v.y; q[2] = (__bf16)v.z; q[3] = (__bf16)v.w;
  }
  __syncthreads();

  const int mi  = (wave >> 2) * 16;
  const int nj0 = (wave & 3) * 8;
  f32x8 acc[8];
#pragma unroll
  for (int t = 0; t < 8; ++t) acc[t] = fzero();

  gemm_pass4<Dm>(xs, LQ, mi, Wb, Dm, nj0 * 16,       lane, acc);
  gemm_pass4<Dm>(xs, LQ, mi, Wb, Dm, (nj0 + 4) * 16, lane, acc + 4);

  const int mrow = mi + ((lane & 16) ? 8 : 0);
#pragma unroll
  for (int t = 0; t < 8; ++t) {
    int n = (nj0 + t) * 16 + (lane & 15);
    float bn = bias[n];
#pragma unroll
    for (int j = 0; j < 8; ++j)
      Y[(m_base + mrow + j) * Dm + n] = (__bf16)(acc[t][j] + bn);
  }
}

// ---------------- fused attention per (b,a,c) tile of 32 query rows ----------------
__global__ void __launch_bounds__(256, 1)
attn_kernel(const float* __restrict__ query,  // [65536, 512] f32
            const __bf16* __restrict__ Wqb,   // [512,512] bf16 [n][k]
            const float* __restrict__ bq,     // [512]
            const __bf16* __restrict__ kproj, // [8,256,512] bf16
            const __bf16* __restrict__ vT,    // [8,512,256] bf16
            float* __restrict__ out) {        // [65536, 512] f32
  extern __shared__ char smem[];
  __bf16* qs   = (__bf16*)smem;                    // [32][LQ]  (aliased by S later)
  __bf16* qp   = (__bf16*)(smem + 32 * LQ * 2);    // [32][LQ]  (aliased by P later)
  float*  S    = (float*)smem;                     // [32][LS]  (reuses qs region)
  __bf16* P    = qp;                               // [32][LP]  (reuses qp region)
  float*  red  = (float*)(smem + 2 * 32 * LQ * 2); // [256]
  float*  rmax = red + 256;                        // [32]
  float*  rsum = rmax + 32;                        // [32]

  const int tid = threadIdx.x, lane = tid & 31, wave = tid >> 5;
  const int blk = blockIdx.x;                      // (b*16 + a)*16 + c
  const int b = blk >> 8;
  const size_t m_base = (size_t)blk * 32;
  const __bf16* kp = kproj + (size_t)b * Tk * Dm;
  const __bf16* vp = vT + (size_t)b * Dm * Tk;

  // stage 0: query tile f32 -> bf16 LDS
  for (int i = tid * 4; i < 32 * Dm; i += 256 * 4) {
    float4 v = *(const float4*)(query + m_base * Dm + i);
    int r = i >> 9, c = i & (Dm - 1);
    __bf16* q = qs + r * LQ + c;
    q[0] = (__bf16)v.x; q[1] = (__bf16)v.y; q[2] = (__bf16)v.z; q[3] = (__bf16)v.w;
  }
  __syncthreads();

  const int mi = (wave >> 2) * 16;

  // stage A: qp = qs @ Wq^T + bq   (32x512, bf16 into LDS)
  {
    const int nj0 = (wave & 3) * 8;
    f32x8 acc[8];
#pragma unroll
    for (int t = 0; t < 8; ++t) acc[t] = fzero();
    gemm_pass4<Dm>(qs, LQ, mi, Wqb, Dm, nj0 * 16,       lane, acc);
    gemm_pass4<Dm>(qs, LQ, mi, Wqb, Dm, (nj0 + 4) * 16, lane, acc + 4);
    const int mrow = mi + ((lane & 16) ? 8 : 0);
#pragma unroll
    for (int t = 0; t < 8; ++t) {
      int n = (nj0 + t) * 16 + (lane & 15);
      float bn = bq[n];
#pragma unroll
      for (int j = 0; j < 8; ++j)
        qp[(mrow + j) * LQ + n] = (__bf16)(acc[t][j] + bn);
    }
  }
  __syncthreads();

  // stage B: S = (qp @ kproj^T) * 1/sqrt(512)   (32x256 f32, overwrites qs)
  {
    const int tj0 = (wave & 3) * 4;
    f32x8 acc[4];
#pragma unroll
    for (int t = 0; t < 4; ++t) acc[t] = fzero();
    gemm_pass4<Dm>(qp, LQ, mi, kp, Dm, tj0 * 16, lane, acc);
    const float scale = 0.044194173824159216f; // 1/sqrt(512)
    const int mrow = mi + ((lane & 16) ? 8 : 0);
#pragma unroll
    for (int t = 0; t < 4; ++t) {
      int n = (tj0 + t) * 16 + (lane & 15);
#pragma unroll
      for (int j = 0; j < 8; ++j)
        S[(mrow + j) * LS + n] = acc[t][j] * scale;
    }
  }
  __syncthreads();

  // stage C: rowwise softmax; unnormalized exp -> bf16 P (overwrites qp),
  // reciprocal row sums -> rsum (folded into the PV accumulators later)
  {
    const int r = tid >> 3;
    const int c0 = (tid & 7) * 32;
    float m = -1e30f;
    for (int i = 0; i < 32; ++i) m = fmaxf(m, S[r * LS + c0 + i]);
    red[tid] = m;
    __syncthreads();
    if ((tid & 7) == 0) {
      float mm = red[tid];
      for (int j = 1; j < 8; ++j) mm = fmaxf(mm, red[tid + j]);
      rmax[r] = mm;
    }
    __syncthreads();
    const float mm = rmax[r];
    float s = 0.f;
    for (int i = 0; i < 32; ++i) {
      float e = __expf(S[r * LS + c0 + i] - mm);
      P[r * LP + c0 + i] = (__bf16)e;
      s += e;
    }
    red[tid] = s;
    __syncthreads();
    if ((tid & 7) == 0) {
      float ss = 0.f;
      for (int j = 0; j < 8; ++j) ss += red[tid + j];
      rsum[r] = 1.0f / ss;
    }
    __syncthreads();
  }

  // stage D: out = diag(1/rowsum) * (P @ V)   (32x512 f32 to global)
  {
    const int nj0 = (wave & 3) * 8;
    f32x8 acc[8];
#pragma unroll
    for (int t = 0; t < 8; ++t) acc[t] = fzero();
    gemm_pass4<Tk>(P, LP, mi, vp, Tk, nj0 * 16,       lane, acc);
    gemm_pass4<Tk>(P, LP, mi, vp, Tk, (nj0 + 4) * 16, lane, acc + 4);
    const int mrow = mi + ((lane & 16) ? 8 : 0);
    float rs[8];
#pragma unroll
    for (int j = 0; j < 8; ++j) rs[j] = rsum[mrow + j];
#pragma unroll
    for (int t = 0; t < 8; ++t) {
      int n = (nj0 + t) * 16 + (lane & 15);
#pragma unroll
      for (int j = 0; j < 8; ++j)
        out[(m_base + mrow + j) * Dm + n] = acc[t][j] * rs[j];
    }
  }
}

extern "C" void kernel_launch(void* const* d_in, const int* in_sizes, int n_in,
                              void* d_out, int out_size, void* d_ws, size_t ws_size,
                              hipStream_t stream) {
  const float* query = (const float*)d_in[0];
  const float* key   = (const float*)d_in[1];
  const float* value = (const float*)d_in[2];
  const float* Wq    = (const float*)d_in[3];
  const float* bq    = (const float*)d_in[4];
  const float* Wk    = (const float*)d_in[5];
  const float* bk    = (const float*)d_in[6];
  float* out = (float*)d_out;

  char* ws = (char*)d_ws;
  __bf16* Wqb   = (__bf16*)(ws);                       // 512 KB
  __bf16* Wkb   = (__bf16*)(ws + (512u << 10));        // 512 KB
  __bf16* kproj = (__bf16*)(ws + (1024u << 10));       // 2 MB  [8,256,512]
  __bf16* vT    = (__bf16*)(ws + (3072u << 10));       // 2 MB  [8,512,256]

  cvt_w_kernel<<<(Dm * Dm + 255) / 256, 256, 0, stream>>>(Wq, Wk, Wqb, Wkb);
  transpose_v_kernel<<<(8 * Tk * Dm + 255) / 256, 256, 0, stream>>>(value, vT);

  // kproj = key @ Wk^T + bk : M = 8*256 = 2048 rows, 32 per WG
  proj_kernel<<<64, 256, 32 * LQ * 2, stream>>>(key, Wkb, bk, kproj);

  // fused attention: one WG per (b,a,c) tile of Q=32 rows
  const size_t attn_smem = 2 * 32 * LQ * 2 + 256 * 4 + 64 * 4;
  attn_kernel<<<2048, 256, attn_smem, stream>>>(query, Wqb, bq, kproj, vT, out);
}
